// Model_56040733278263
// MI455X (gfx1250) — matrix operand order repaired and checked by reading, not compiled, not run on hardware
//
#include <hip/hip_runtime.h>

typedef __attribute__((ext_vector_type(16))) _Float16 v16h;
typedef __attribute__((ext_vector_type(8)))  _Float16 v8h;
typedef __attribute__((ext_vector_type(8)))  float    v8f;

#define EPSV 1e-5f

// ---------------------------------------------------------------------------
// Layer 1: direct conv (C_in = 1, C_out = 16). K=1 would waste a WMMA, so VALU.
// Input: fp32 NCHW [2,1,H,W]. Output: raw (pre-BN) f16 NHWC with Cpad=32,
// channels 0..15 written (16..31 are zero-filled later by bn_relu).
// ---------------------------------------------------------------------------
__global__ void conv1_direct(const float* __restrict__ x, const float* __restrict__ w,
                             _Float16* __restrict__ out, int H, int W) {
  __shared__ float ws[144];
  int t = threadIdx.x;
  if (t < 144) ws[t] = w[t];            // w1: [16][1][3][3]
  __syncthreads();
  long long idx   = (long long)blockIdx.x * blockDim.x + t;
  long long total = 2LL * H * W;
  if (idx >= total) return;
  int xw = (int)(idx % W);
  int y  = (int)((idx / W) % H);
  int n  = (int)(idx / ((long long)W * H));
  const float* xp = x + (long long)n * H * W;
  float p[9];
#pragma unroll
  for (int ky = 0; ky < 3; ++ky)
#pragma unroll
    for (int kx = 0; kx < 3; ++kx) {
      int yy = y + ky - 1, xx = xw + kx - 1;
      p[ky * 3 + kx] = (yy >= 0 && yy < H && xx >= 0 && xx < W)
                           ? xp[(long long)yy * W + xx] : 0.f;
    }
  v16h o;
#pragma unroll
  for (int oc = 0; oc < 16; ++oc) {
    float s = 0.f;
#pragma unroll
    for (int k = 0; k < 9; ++k) s += p[k] * ws[oc * 9 + k];
    o[oc] = (_Float16)s;
  }
  *(v16h*)(out + idx * 32) = o;         // channels 0..15 of Cpad=32 pixel
}

// ---------------------------------------------------------------------------
// Pack fp32 OIHW weights into a contiguous-k WMMA A-operand order (f16 16x32), matching B's order:
//   element (m,k): lane = m + 16*(k/16), vgpr = (k%16)/2, half = k%2
// Packed index: (((mt*9 + tap)*chunks + ch)*32 + lane)*16 + (vgpr*2+half)
// Zero-pads ic >= Cin (channel padding) so GEMM over padded K is exact.
// ---------------------------------------------------------------------------
__global__ void pack_weights(const float* __restrict__ w, _Float16* __restrict__ pk,
                             int Cin, int Cout, int chunks, int total) {
  int idx = blockIdx.x * blockDim.x + threadIdx.x;
  if (idx >= total) return;
  int h    = idx & 15;
  int lane = (idx >> 4) & 31;
  int ch   = (idx >> 9) % chunks;
  int rest = (idx >> 9) / chunks;
  int tap  = rest % 9;
  int mt   = rest / 9;
  int vgpr = h >> 1, half = h & 1;
  int m  = lane & 15;
  int hi = lane >> 4;
  int k  = hi * 16 + vgpr * 2 + half;
  int ic = ch * 32 + k;
  int oc = mt * 16 + m;
  float val = (ic < Cin && oc < Cout) ? w[((long long)oc * Cin + ic) * 9 + tap] : 0.f;
  pk[idx] = (_Float16)val;
}

// ---------------------------------------------------------------------------
// Implicit-GEMM 3x3 SAME conv via v_wmma_f32_16x16x32_f16.
// Grid: (ceil(tilesX/4), H, N); block = 128 (4 waves). Each wave computes a
// 16-pixel row segment x all Cout tiles (MTILES accumulators).
//   B (32x16, K=chan, N=pixel): lane = (pix%16) + 16*(k/16); its 16 halves are
//     16 contiguous channels at that pixel  -> single 32B v16h load (NHWC).
//   A: prepacked (above), v16h per lane (two b128 loads).
//   C/D 16x16 f32: (m,n) -> vgpr m%8, lane n + 16*(m/8); the 8 f32 per lane
//     convert to one contiguous v8h store (NHWC, oc-contiguous).
// Spatial SAME padding + x-tile overhang handled by predicated (zeroed) B loads.
// The tx guard is wave-uniform, so EXEC is all-ones around every WMMA.
// ---------------------------------------------------------------------------
template <int CHUNKS, int MTILES>
__global__ void conv_wmma(const _Float16* __restrict__ in, const _Float16* __restrict__ wp,
                          _Float16* __restrict__ out, int H, int W, int CpadOut, int tilesX) {
  const int CpadIn = CHUNKS * 32;
  int lane = threadIdx.x & 31;
  int wv   = threadIdx.x >> 5;
  int tx   = blockIdx.x * 4 + wv;
  if (tx >= tilesX) return;           // wave-uniform
  int y    = blockIdx.y;
  int n    = blockIdx.z;
  int nIdx = lane & 15;   // pixel within 16-wide tile (GEMM N)
  int kgrp = lane >> 4;   // K half-group selector
  int xs   = tx * 16 + nIdx;

  v8f acc[MTILES];
#pragma unroll
  for (int mt = 0; mt < MTILES; ++mt) acc[mt] = (v8f){};

#pragma unroll
  for (int tap = 0; tap < 9; ++tap) {
    int dy = tap / 3 - 1, dx = tap % 3 - 1;
    int yy = y + dy, xx = xs + dx;
    bool pv = (yy >= 0) && (yy < H) && (xx >= 0) && (xx < W);
    long long pixbase =
        ((long long)(n * H + (pv ? yy : 0)) * W + (pv ? xx : 0)) * CpadIn;
#pragma unroll
    for (int ch = 0; ch < CHUNKS; ++ch) {
      v16h b = {};
      if (pv) b = *(const v16h*)(in + pixbase + ch * 32 + kgrp * 16);
      const _Float16* ap = wp + ((long long)(tap * CHUNKS + ch) * 32 + lane) * 16;
#pragma unroll
      for (int mt = 0; mt < MTILES; ++mt) {
        v16h a = *(const v16h*)(ap + (long long)mt * 9 * CHUNKS * 512);
        acc[mt] = __builtin_amdgcn_wmma_f32_16x16x32_f16(
            false, a, false, b, (short)0, acc[mt], false, false);
      }
    }
  }

  if (xs < W) {
    long long obase = ((long long)(n * H + y) * W + xs) * CpadOut;
#pragma unroll
    for (int mt = 0; mt < MTILES; ++mt) {
      v8h o;
#pragma unroll
      for (int v = 0; v < 8; ++v) o[v] = (_Float16)acc[mt][v];
      *(v8h*)(out + obase + mt * 16 + kgrp * 8) = o;
    }
  }
}

// ---------------------------------------------------------------------------
// Per-channel sum / sum-of-squares over (N,H,W) of raw f16 NHWC conv output.
// sums[0..95] = sum, sums[96..191] = sumsq.
// ---------------------------------------------------------------------------
__global__ void chan_stats(const _Float16* __restrict__ buf, float* __restrict__ sums,
                           int Cpad, long long P) {
  __shared__ float s1[256], s2[256];
  int c = blockIdx.y;
  long long base = (long long)blockIdx.x * 256 * 16 + threadIdx.x;
  float a = 0.f, b = 0.f;
  for (int i = 0; i < 16; ++i) {
    long long p = base + (long long)i * 256;
    if (p < P) { float v = (float)buf[p * Cpad + c]; a += v; b += v * v; }
  }
  s1[threadIdx.x] = a; s2[threadIdx.x] = b;
  __syncthreads();
  for (int off = 128; off > 0; off >>= 1) {
    if (threadIdx.x < off) {
      s1[threadIdx.x] += s1[threadIdx.x + off];
      s2[threadIdx.x] += s2[threadIdx.x + off];
    }
    __syncthreads();
  }
  if (threadIdx.x == 0) {
    atomicAdd(&sums[c], s1[0]);
    atomicAdd(&sums[96 + c], s2[0]);
  }
}

__global__ void bn_finalize(const float* __restrict__ sums, const float* __restrict__ gb,
                            float* __restrict__ scale, float* __restrict__ shift,
                            int Cout, float invP) {
  int c = threadIdx.x;
  if (c < Cout) {
    float mu  = sums[c] * invP;
    float var = sums[96 + c] * invP - mu * mu;
    float sc  = gb[c] * rsqrtf(var + EPSV);
    scale[c]  = sc;
    shift[c]  = gb[Cout + c] - mu * sc;
  }
}

// In-place BN+ReLU; also writes 0 to padded channels so next GEMM's K-padding
// invariant holds (and maxpool of pads stays 0).
__global__ void bn_relu(_Float16* __restrict__ buf, const float* __restrict__ scale,
                        const float* __restrict__ shift, int Cout, int Cpad,
                        long long total) {
  long long idx = (long long)blockIdx.x * blockDim.x + threadIdx.x;
  if (idx >= total) return;
  int c = (int)(idx % Cpad);
  float v = 0.f;
  if (c < Cout) {
    v = (float)buf[idx] * scale[c] + shift[c];
    v = v > 0.f ? v : 0.f;
  }
  buf[idx] = (_Float16)v;
}

// 3x3 stride-2 VALID maxpool, NHWC. Inputs are post-ReLU (>=0), pads are 0,
// and (Ho-1)*2+2 == Hi-1 so every window is fully in-bounds.
__global__ void maxpool3s2(const _Float16* __restrict__ in, _Float16* __restrict__ out,
                           int Hi, int Wi, int Ho, int Wo, int Cpad, long long total) {
  long long idx = (long long)blockIdx.x * blockDim.x + threadIdx.x;
  if (idx >= total) return;
  int c = (int)(idx % Cpad);
  long long r = idx / Cpad;
  int wo = (int)(r % Wo); r /= Wo;
  int ho = (int)(r % Ho);
  int n  = (int)(r / Ho);
  float m = 0.f;
#pragma unroll
  for (int i = 0; i < 3; ++i)
#pragma unroll
    for (int j = 0; j < 3; ++j) {
      float v = (float)in[((long long)(n * Hi + ho * 2 + i) * Wi + (wo * 2 + j)) * Cpad + c];
      m = fmaxf(m, v);
    }
  out[idx] = (_Float16)m;
}

// Mean over 100x100 spatial -> feat[n*80+c]; one block per (n,c).
__global__ void mean_hw(const _Float16* __restrict__ buf, float* __restrict__ feat, int Cpad) {
  __shared__ float sd[256];
  int c = blockIdx.x % 80;
  int n = blockIdx.x / 80;
  float s = 0.f;
  for (int p = threadIdx.x; p < 10000; p += 256)
    s += (float)buf[((long long)(n * 10000 + p)) * Cpad + c];
  sd[threadIdx.x] = s;
  __syncthreads();
  for (int off = 128; off > 0; off >>= 1) {
    if (threadIdx.x < off) sd[threadIdx.x] += sd[threadIdx.x + off];
    __syncthreads();
  }
  if (threadIdx.x == 0) feat[n * 80 + c] = sd[0] * 1e-4f;
}

// BN1d over batch of 2 (batch stats) + linear 80 -> 40, fp32 out [2,40].
__global__ void head(const float* __restrict__ feat, const float* __restrict__ bn_gb,
                     const float* __restrict__ lin_w, const float* __restrict__ lin_b,
                     float* __restrict__ outp) {
  __shared__ float hn[160];
  int t = threadIdx.x;
  if (t < 80) {
    float f0 = feat[t], f1 = feat[80 + t];
    float mu  = 0.5f * (f0 + f1);
    float var = 0.5f * ((f0 - mu) * (f0 - mu) + (f1 - mu) * (f1 - mu));
    float r   = rsqrtf(var + EPSV);
    hn[t]      = (f0 - mu) * r * bn_gb[t] + bn_gb[80 + t];
    hn[80 + t] = (f1 - mu) * r * bn_gb[t] + bn_gb[80 + t];
  }
  __syncthreads();
  if (t < 80) {
    int n = t / 40, j = t % 40;
    float s = lin_b[j];
    for (int c = 0; c < 80; ++c) s += hn[n * 80 + c] * lin_w[j * 80 + c];
    outp[t] = s;
  }
}

// ---------------------------------------------------------------------------
// Host driver
// ---------------------------------------------------------------------------
static void bn_stage(_Float16* buf, const float* gb, int Cout, int Cpad, int H, int W,
                     float* sums, float* scale, float* shiftv, hipStream_t stream) {
  long long P = 2LL * H * W;
  hipMemsetAsync(sums, 0, 192 * sizeof(float), stream);
  int pixBlocks = (int)((P + 4095) / 4096);
  chan_stats<<<dim3(pixBlocks, Cout), 256, 0, stream>>>(buf, sums, Cpad, P);
  bn_finalize<<<1, 96, 0, stream>>>(sums, gb, scale, shiftv, Cout, (float)(1.0 / (double)P));
  long long tot = P * Cpad;
  bn_relu<<<(int)((tot + 255) / 256), 256, 0, stream>>>(buf, scale, shiftv, Cout, Cpad, tot);
}

static inline dim3 conv_grid(int tilesX, int H) {
  return dim3((unsigned)((tilesX + 3) / 4), (unsigned)H, 2u);
}

extern "C" void kernel_launch(void* const* d_in, const int* in_sizes, int n_in,
                              void* d_out, int out_size, void* d_ws, size_t ws_size,
                              hipStream_t stream) {
  (void)in_sizes; (void)n_in; (void)out_size; (void)ws_size;

  const float* x = (const float*)d_in[0];
  const float* w[11];
  const float* gb[11];
  for (int i = 1; i <= 10; ++i) {
    w[i]  = (const float*)d_in[i];
    gb[i] = (const float*)d_in[10 + i];
  }
  const float* bn_gb = (const float*)d_in[21];
  const float* lin_w = (const float*)d_in[22];
  const float* lin_b = (const float*)d_in[23];

  // workspace layout (f16 ping-pong activations, packed weights, stats)
  const size_t SZ1 = (size_t)2 * 32 * 1615 * 1615;   // elements per big buffer
  char* base = (char*)d_ws;
  _Float16* A  = (_Float16*)base;
  _Float16* B  = (_Float16*)(base + SZ1 * 2);
  _Float16* wp = (_Float16*)(base + SZ1 * 4);
  float* stats  = (float*)(base + SZ1 * 4 + (1 << 20));
  float* sums   = stats;         // [192]
  float* scale  = stats + 192;   // [96]
  float* shiftv = stats + 288;   // [96]
  float* feat   = stats + 384;   // [160]

  // ---- pack WMMA A-operand weights for conv2..conv10 ----
  const int Cin_[11]  = {0, 1, 16, 16, 32, 32, 48, 48, 64, 64, 80};
  const int Cout_[11] = {0, 16, 16, 32, 32, 48, 48, 64, 64, 80, 80};
  const int ch_[11]   = {0, 0, 1, 1, 1, 1, 2, 2, 2, 2, 3};
  const int mt_[11]   = {0, 0, 1, 2, 2, 3, 3, 4, 4, 5, 5};
  size_t woff[12];
  woff[2] = 0;
  for (int l = 2; l <= 10; ++l) {
    int total = mt_[l] * 9 * ch_[l] * 512;
    pack_weights<<<(total + 255) / 256, 256, 0, stream>>>(
        w[l], wp + woff[l], Cin_[l], Cout_[l], ch_[l], total);
    woff[l + 1] = woff[l] + (size_t)total;
  }

  // ---- block 0 @1615 ----
  {
    long long tot = 2LL * 1615 * 1615;
    conv1_direct<<<(int)((tot + 255) / 256), 256, 0, stream>>>(x, w[1], A, 1615, 1615);
    bn_stage(A, gb[1], 16, 32, 1615, 1615, sums, scale, shiftv, stream);
  }
  {
    conv_wmma<1, 1><<<conv_grid(101, 1615), 128, 0, stream>>>(A, wp + woff[2], B, 1615, 1615, 32, 101);
    bn_stage(B, gb[2], 16, 32, 1615, 1615, sums, scale, shiftv, stream);
    long long tot = 2LL * 807 * 807 * 32;
    maxpool3s2<<<(int)((tot + 255) / 256), 256, 0, stream>>>(B, A, 1615, 1615, 807, 807, 32, tot);
  }
  // ---- block 1 @807 ----
  {
    conv_wmma<1, 2><<<conv_grid(51, 807), 128, 0, stream>>>(A, wp + woff[3], B, 807, 807, 32, 51);
    bn_stage(B, gb[3], 32, 32, 807, 807, sums, scale, shiftv, stream);
  }
  {
    conv_wmma<1, 2><<<conv_grid(51, 807), 128, 0, stream>>>(B, wp + woff[4], A, 807, 807, 32, 51);
    bn_stage(A, gb[4], 32, 32, 807, 807, sums, scale, shiftv, stream);
    long long tot = 2LL * 403 * 403 * 32;
    maxpool3s2<<<(int)((tot + 255) / 256), 256, 0, stream>>>(A, B, 807, 807, 403, 403, 32, tot);
  }
  // ---- block 2 @403 ----
  {
    conv_wmma<1, 3><<<conv_grid(26, 403), 128, 0, stream>>>(B, wp + woff[5], A, 403, 403, 64, 26);
    bn_stage(A, gb[5], 48, 64, 403, 403, sums, scale, shiftv, stream);
  }
  {
    conv_wmma<2, 3><<<conv_grid(26, 403), 128, 0, stream>>>(A, wp + woff[6], B, 403, 403, 64, 26);
    bn_stage(B, gb[6], 48, 64, 403, 403, sums, scale, shiftv, stream);
    long long tot = 2LL * 201 * 201 * 64;
    maxpool3s2<<<(int)((tot + 255) / 256), 256, 0, stream>>>(B, A, 403, 403, 201, 201, 64, tot);
  }
  // ---- block 3 @201 ----
  {
    conv_wmma<2, 4><<<conv_grid(13, 201), 128, 0, stream>>>(A, wp + woff[7], B, 201, 201, 64, 13);
    bn_stage(B, gb[7], 64, 64, 201, 201, sums, scale, shiftv, stream);
  }
  {
    conv_wmma<2, 4><<<conv_grid(13, 201), 128, 0, stream>>>(B, wp + woff[8], A, 201, 201, 64, 13);
    bn_stage(A, gb[8], 64, 64, 201, 201, sums, scale, shiftv, stream);
    long long tot = 2LL * 100 * 100 * 64;
    maxpool3s2<<<(int)((tot + 255) / 256), 256, 0, stream>>>(A, B, 201, 201, 100, 100, 64, tot);
  }
  // ---- block 4 @100 ----
  {
    conv_wmma<2, 5><<<conv_grid(7, 100), 128, 0, stream>>>(B, wp + woff[9], A, 100, 100, 96, 7);
    bn_stage(A, gb[9], 80, 96, 100, 100, sums, scale, shiftv, stream);
  }
  {
    conv_wmma<3, 5><<<conv_grid(7, 100), 128, 0, stream>>>(A, wp + woff[10], B, 100, 100, 96, 7);
    bn_stage(B, gb[10], 80, 96, 100, 100, sums, scale, shiftv, stream);
  }
  // ---- head ----
  mean_hw<<<160, 256, 0, stream>>>(B, feat, 96);
  head<<<1, 128, 0, stream>>>(feat, bn_gb, lin_w, lin_b, (float*)d_out);
}
